// SelfAttention_88433376624716
// MI455X (gfx1250) — matrix operand order repaired and checked by reading, not compiled, not run on hardware
//
#include <hip/hip_runtime.h>
#include <hip/hip_bf16.h>

typedef __attribute__((ext_vector_type(16))) _Float16 v16h;
typedef __attribute__((ext_vector_type(8)))  _Float16 v8h;
typedef __attribute__((ext_vector_type(8)))  float    v8f;

#define B_   8
#define C_   512
#define L_   4096
#define C3_  1536
#define GC_  16          // channels per group (512/32)
#define EPS_ 1e-5f

// ---------------------------------------------------------------- WMMA utils

__device__ __forceinline__ v8f wmma_f16(v16h a, v16h b, v8f c) {
  // D = A(16x32 f16) x B(32x16 f16) + C(16x16 f32)
  return __builtin_amdgcn_wmma_f32_16x16x32_f16(false, a, false, b,
                                                (short)0, c, false, false);
}

__device__ __forceinline__ v16h make16(v8h lo, v8h hi) {
  return __builtin_shufflevector(lo, hi, 0,1,2,3,4,5,6,7,8,9,10,11,12,13,14,15);
}

// A fragment (16x32 f16, row-major source): lane holds row (lane&15),
// khalf = lane>>4; vector elems 0..15 = K khalf*16+0..15
__device__ __forceinline__ v16h load_a(const _Float16* rowbase, int khalf) {
  return *(const v16h*)(rowbase + khalf * 16);
}

// ---------------------------------------------------------------- K0: weights -> f16

__global__ void k_convert_w(const float* __restrict__ qkv_w,
                            const float* __restrict__ out_w,
                            _Float16* __restrict__ wq,
                            _Float16* __restrict__ wo) {
  int i = blockIdx.x * 256 + threadIdx.x;
  if (i < C3_ * C_) wq[i] = (_Float16)qkv_w[i];
  if (i < C_ * C_)  wo[i] = (_Float16)out_w[i];
}

// ---------------------------------------------------------------- K1: GroupNorm -> xnT[L,C] f16

__global__ void k_groupnorm(const float* __restrict__ x,
                            const float* __restrict__ gamma,
                            const float* __restrict__ beta,
                            _Float16* __restrict__ xnT) {
  __shared__ float ssum[256], ssq[256], sstat[2];
  int bg = blockIdx.x;                 // b*32 + g
  int b = bg >> 5, g = bg & 31;
  const float* xb = x + (size_t)(b * C_ + g * GC_) * L_;
  int t = threadIdx.x;
  float sum = 0.f, sq = 0.f;
  for (int i = t; i < GC_ * L_; i += 256) {
    float v = xb[i];
    sum += v; sq += v * v;
  }
  ssum[t] = sum; ssq[t] = sq;
  __syncthreads();
  for (int off = 128; off > 0; off >>= 1) {
    if (t < off) { ssum[t] += ssum[t + off]; ssq[t] += ssq[t + off]; }
    __syncthreads();
  }
  if (t == 0) {
    float inv_n = 1.0f / (float)(GC_ * L_);
    float mean = ssum[0] * inv_n;
    float var  = ssq[0] * inv_n - mean * mean;
    sstat[0] = mean;
    sstat[1] = rsqrtf(var + EPS_);
  }
  __syncthreads();
  float mean = sstat[0], rstd = sstat[1];
  _Float16* xout = xnT + (size_t)b * L_ * C_;
  for (int i = t; i < GC_ * L_; i += 256) {
    int cl = i >> 12;                  // i / 4096
    int l  = i & (L_ - 1);
    int c  = g * GC_ + cl;
    float v = (xb[i] - mean) * rstd * gamma[c] + beta[c];
    xout[(size_t)l * C_ + c] = (_Float16)v;
  }
}

// ---------------------------------------------------------------- K2: QKV GEMM
// qkv[o,l] = sum_c wq[o,c] * xnT[l,c]  (per batch). wave: 16o x 64l tile.
// q -> qT[L,C] (scaled), k -> kT[L,C], v -> v[C,L]

__global__ void k_qkv_gemm(const _Float16* __restrict__ wq,
                           const float* __restrict__ qkv_b,
                           const _Float16* __restrict__ xnT,
                           _Float16* __restrict__ qT,
                           _Float16* __restrict__ kT,
                           _Float16* __restrict__ vv) {
  const float scale = 0.044194173824159216f;   // 1/sqrt(512)
  int wid  = blockIdx.x * 8 + (threadIdx.x >> 5);
  int lane = threadIdx.x & 31;
  int col = lane & 15, khalf = lane >> 4;
  int b  = wid / (96 * 64);
  int r  = wid % (96 * 64);
  int o0 = (r / 64) * 16;
  int l0 = (r % 64) * 64;
  const _Float16* xb = xnT + (size_t)b * L_ * C_;

  v8f acc0 = {}, acc1 = {}, acc2 = {}, acc3 = {};
  for (int kc = 0; kc < 16; ++kc) {
    int c0 = kc * 32;
    v16h a = load_a(wq + (size_t)(o0 + col) * C_ + c0, khalf);
    const _Float16* bb = xb + (size_t)(l0 + col) * C_ + c0 + khalf * 16;
    acc0 = wmma_f16(a, *(const v16h*)(bb),           acc0);
    acc1 = wmma_f16(a, *(const v16h*)(bb + 16 * C_), acc1);
    acc2 = wmma_f16(a, *(const v16h*)(bb + 32 * C_), acc2);
    acc3 = wmma_f16(a, *(const v16h*)(bb + 48 * C_), acc3);
  }

  int orow0 = o0 + khalf * 8;
  _Float16* qb = qT + (size_t)b * L_ * C_;
  _Float16* kb = kT + (size_t)b * L_ * C_;
  _Float16* vb = vv + (size_t)b * C_ * L_;
#pragma unroll
  for (int e = 0; e < 8; ++e) {
    int orow = orow0 + e;
    float bias = qkv_b[orow];
    float v0 = acc0[e] + bias, v1 = acc1[e] + bias;
    float v2 = acc2[e] + bias, v3 = acc3[e] + bias;
    int l = l0 + col;
    if (orow < 512) {
      qb[(size_t)(l     ) * C_ + orow] = (_Float16)(v0 * scale);
      qb[(size_t)(l + 16) * C_ + orow] = (_Float16)(v1 * scale);
      qb[(size_t)(l + 32) * C_ + orow] = (_Float16)(v2 * scale);
      qb[(size_t)(l + 48) * C_ + orow] = (_Float16)(v3 * scale);
    } else if (orow < 1024) {
      int o = orow - 512;
      kb[(size_t)(l     ) * C_ + o] = (_Float16)v0;
      kb[(size_t)(l + 16) * C_ + o] = (_Float16)v1;
      kb[(size_t)(l + 32) * C_ + o] = (_Float16)v2;
      kb[(size_t)(l + 48) * C_ + o] = (_Float16)v3;
    } else {
      int o = orow - 1024;
      vb[(size_t)o * L_ + l     ] = (_Float16)v0;
      vb[(size_t)o * L_ + l + 16] = (_Float16)v1;
      vb[(size_t)o * L_ + l + 32] = (_Float16)v2;
      vb[(size_t)o * L_ + l + 48] = (_Float16)v3;
    }
  }
}

// ---------------------------------------------------------------- K3a: softmax stats
// Per wave: 16 query rows, stream all K tiles, online max / sumexp.

__global__ void k_attn_stats(const _Float16* __restrict__ qT,
                             const _Float16* __restrict__ kT,
                             float* __restrict__ stats) {
  int wid  = blockIdx.x * 8 + (threadIdx.x >> 5);
  int lane = threadIdx.x & 31;
  int col = lane & 15, khalf = lane >> 4;
  int b  = wid >> 8;                 // / 256
  int l0 = (wid & 255) * 16;
  const _Float16* qb = qT + (size_t)b * L_ * C_;
  const _Float16* kb = kT + (size_t)b * L_ * C_;

  v16h qa[16];
#pragma unroll
  for (int kc = 0; kc < 16; ++kc)
    qa[kc] = load_a(qb + (size_t)(l0 + col) * C_ + kc * 32, khalf);

  float m[8], ssum[8];
#pragma unroll
  for (int e = 0; e < 8; ++e) { m[e] = -1e30f; ssum[e] = 0.f; }

  for (int st = 0; st < 256; ++st) {
    int s0 = st * 16;
    v8f acc = {};
#pragma unroll
    for (int kc = 0; kc < 16; ++kc) {
      v16h bf = *(const v16h*)(kb + (size_t)(s0 + col) * C_ + kc * 32 + khalf * 16);
      acc = wmma_f16(qa[kc], bf, acc);
    }
#pragma unroll
    for (int e = 0; e < 8; ++e) {
      float r = acc[e];
      float rm = r;
      rm = fmaxf(rm, __shfl_xor(rm, 1));
      rm = fmaxf(rm, __shfl_xor(rm, 2));
      rm = fmaxf(rm, __shfl_xor(rm, 4));
      rm = fmaxf(rm, __shfl_xor(rm, 8));        // row max across 16 cols
      float newm = fmaxf(m[e], rm);
      float p = __expf(r - newm);
      float rs = p;
      rs += __shfl_xor(rs, 1);
      rs += __shfl_xor(rs, 2);
      rs += __shfl_xor(rs, 4);
      rs += __shfl_xor(rs, 8);                  // row sum across 16 cols
      ssum[e] = ssum[e] * __expf(m[e] - newm) + rs;
      m[e] = newm;
    }
  }
  if (col == 0) {
#pragma unroll
    for (int e = 0; e < 8; ++e) {
      int row = l0 + khalf * 8 + e;
      stats[2 * ((size_t)b * L_ + row) + 0] = m[e];
      stats[2 * ((size_t)b * L_ + row) + 1] = ssum[e];
    }
  }
}

// ---------------------------------------------------------------- K3b: O = softmax(S) @ V^T
// Per wave: 16 query rows, LDS f32 accumulator 16x512, LDS f16 P-staging 16x32.

#define OACC_BYTES (16 * 512 * 4)
#define STG_BYTES  (16 * 32 * 2)
#define WAVE_BYTES (OACC_BYTES + STG_BYTES)   // 33792

__global__ void k_attn_pv(const _Float16* __restrict__ qT,
                          const _Float16* __restrict__ kT,
                          const _Float16* __restrict__ vv,
                          const float* __restrict__ stats,
                          _Float16* __restrict__ oT) {
  extern __shared__ char smem[];
  int wave = threadIdx.x >> 5;
  int lane = threadIdx.x & 31;
  int col = lane & 15, khalf = lane >> 4;
  int b  = blockIdx.x >> 6;                    // / 64
  int l0 = (blockIdx.x & 63) * 64 + wave * 16;

  float*    Oacc = (float*)(smem + wave * WAVE_BYTES);
  _Float16* stg  = (_Float16*)(smem + wave * WAVE_BYTES + OACC_BYTES);

  for (int i = lane; i < 16 * 512; i += 32) Oacc[i] = 0.f;

  const _Float16* qb = qT + (size_t)b * L_ * C_;
  const _Float16* kb = kT + (size_t)b * L_ * C_;
  const _Float16* vb = vv + (size_t)b * C_ * L_;

  v16h qa[16];
#pragma unroll
  for (int kc = 0; kc < 16; ++kc)
    qa[kc] = load_a(qb + (size_t)(l0 + col) * C_ + kc * 32, khalf);

  float mrow[8], rinv[8];
#pragma unroll
  for (int e = 0; e < 8; ++e) {
    int row = l0 + khalf * 8 + e;
    mrow[e] = stats[2 * ((size_t)b * L_ + row) + 0];
    rinv[e] = 1.0f / stats[2 * ((size_t)b * L_ + row) + 1];
  }

  for (int sc = 0; sc < 128; ++sc) {
    int s0 = sc * 32;
    // scores for two 16-wide s tiles -> P staged as f16 in LDS (A layout source)
#pragma unroll
    for (int t = 0; t < 2; ++t) {
      v8f acc = {};
#pragma unroll
      for (int kc = 0; kc < 16; ++kc) {
        v16h bf = *(const v16h*)(kb + (size_t)(s0 + t * 16 + col) * C_ + kc * 32 + khalf * 16);
        acc = wmma_f16(qa[kc], bf, acc);
      }
#pragma unroll
      for (int e = 0; e < 8; ++e) {
        float p = __expf(acc[e] - mrow[e]) * rinv[e];
        stg[(khalf * 8 + e) * 32 + t * 16 + col] = (_Float16)p;
      }
    }
    // read P back in A-fragment layout (wave-private LDS, in-order DS ops)
    v16h pa = load_a(stg + col * 32, khalf);
    // O[l, c] += P[16x32] x V^T[32x16] over all 512 channels
#pragma unroll 4
    for (int ct = 0; ct < 32; ++ct) {
      int c0 = ct * 16;
      v16h bf = *(const v16h*)(vb + (size_t)(c0 + col) * L_ + s0 + khalf * 16);
      v8f z = {};
      v8f d = wmma_f16(pa, bf, z);
#pragma unroll
      for (int e = 0; e < 8; ++e)
        Oacc[(khalf * 8 + e) * 512 + c0 + col] += d[e];
    }
  }

  _Float16* ob = oT + (size_t)b * L_ * C_ + (size_t)l0 * C_;
  for (int i = lane; i < 16 * 512; i += 32)
    ob[i] = (_Float16)Oacc[i];   // contiguous since C_ == 512
}

// ---------------------------------------------------------------- K4: out projection + residual

__global__ void k_out_proj(const _Float16* __restrict__ wo,
                           const float* __restrict__ out_b,
                           const _Float16* __restrict__ oT,
                           const float* __restrict__ x,
                           float* __restrict__ out) {
  int wid  = blockIdx.x * 8 + (threadIdx.x >> 5);
  int lane = threadIdx.x & 31;
  int col = lane & 15, khalf = lane >> 4;
  int b  = wid / (32 * 64);
  int r  = wid % (32 * 64);
  int o0 = (r / 64) * 16;
  int l0 = (r % 64) * 64;
  const _Float16* ob = oT + (size_t)b * L_ * C_;

  v8f acc0 = {}, acc1 = {}, acc2 = {}, acc3 = {};
  for (int kc = 0; kc < 16; ++kc) {
    int c0 = kc * 32;
    v16h a = load_a(wo + (size_t)(o0 + col) * C_ + c0, khalf);
    const _Float16* bb = ob + (size_t)(l0 + col) * C_ + c0 + khalf * 16;
    acc0 = wmma_f16(a, *(const v16h*)(bb),           acc0);
    acc1 = wmma_f16(a, *(const v16h*)(bb + 16 * C_), acc1);
    acc2 = wmma_f16(a, *(const v16h*)(bb + 32 * C_), acc2);
    acc3 = wmma_f16(a, *(const v16h*)(bb + 48 * C_), acc3);
  }

#pragma unroll
  for (int e = 0; e < 8; ++e) {
    int orow = o0 + khalf * 8 + e;
    float bias = out_b[orow];
    const float* xr = x   + ((size_t)b * C_ + orow) * L_;
    float*       op = out + ((size_t)b * C_ + orow) * L_;
    int l = l0 + col;
    op[l     ] = acc0[e] + bias + xr[l     ];
    op[l + 16] = acc1[e] + bias + xr[l + 16];
    op[l + 32] = acc2[e] + bias + xr[l + 32];
    op[l + 48] = acc3[e] + bias + xr[l + 48];
  }
}

// ---------------------------------------------------------------- launch

extern "C" void kernel_launch(void* const* d_in, const int* in_sizes, int n_in,
                              void* d_out, int out_size, void* d_ws, size_t ws_size,
                              hipStream_t stream) {
  const float* x      = (const float*)d_in[0];
  const float* norm_w = (const float*)d_in[1];
  const float* norm_b = (const float*)d_in[2];
  const float* qkv_w  = (const float*)d_in[3];
  const float* qkv_b  = (const float*)d_in[4];
  const float* out_w  = (const float*)d_in[5];
  const float* out_b  = (const float*)d_in[6];
  float* out = (float*)d_out;

  char* ws = (char*)d_ws;
  const size_t SZ_WQ  = (size_t)C3_ * C_ * 2;        // 1,572,864
  const size_t SZ_WO  = (size_t)C_ * C_ * 2;         //   524,288
  const size_t SZ_ACT = (size_t)B_ * L_ * C_ * 2;    // 33,554,432
  _Float16* wq  = (_Float16*)(ws);
  _Float16* wo  = (_Float16*)(ws + SZ_WQ);
  _Float16* xnT = (_Float16*)(ws + SZ_WQ + SZ_WO);               // reused as oT
  _Float16* qT  = (_Float16*)(ws + SZ_WQ + SZ_WO + SZ_ACT);
  _Float16* kT  = (_Float16*)(ws + SZ_WQ + SZ_WO + 2 * SZ_ACT);
  _Float16* vv  = (_Float16*)(ws + SZ_WQ + SZ_WO + 3 * SZ_ACT);
  float*    st  = (float*)   (ws + SZ_WQ + SZ_WO + 4 * SZ_ACT);  // [B,L] x {max,sum}

  k_convert_w <<<3072, 256, 0, stream>>>(qkv_w, out_w, wq, wo);
  k_groupnorm <<< 256, 256, 0, stream>>>(x, norm_w, norm_b, xnT);
  k_qkv_gemm  <<<6144, 256, 0, stream>>>(wq, qkv_b, xnT, qT, kT, vv);
  k_attn_stats<<< 256, 256, 0, stream>>>(qT, kT, st);
  k_attn_pv   <<< 512, 128, 4 * WAVE_BYTES, stream>>>(qT, kT, vv, st, xnT /* oT */);
  k_out_proj  <<<2048, 256, 0, stream>>>(wo, out_b, xnT /* oT */, x, out);
}
